// GATLayer_42296837931388
// MI455X (gfx1250) — matrix operand order repaired and checked by reading, not compiled, not run on hardware
//
#include <hip/hip_runtime.h>
#include <hip/hip_bf16.h>

#define B_ 8
#define N_ 1024
#define D_ 256
#define H_ 8
#define F_ 32

typedef __attribute__((ext_vector_type(16))) __bf16       v16bf;
typedef __attribute__((ext_vector_type(8)))  float        v8f;
typedef __attribute__((ext_vector_type(4)))  unsigned int v4u;
typedef __attribute__((ext_vector_type(8)))  int          v8i;
typedef __attribute__((ext_vector_type(4)))  int          v4i;

union BFrag { v16bf v; unsigned short u[16]; unsigned int d[8]; };

__device__ __forceinline__ unsigned short f2bf(float f) {
  union { float f; unsigned u; } x; x.f = f;
  unsigned u = x.u;
  if ((u & 0x7F800000u) != 0x7F800000u)      // not inf/nan: round-to-nearest-even
    u = u + 0x7FFFu + ((u >> 16) & 1u);
  return (unsigned short)(u >> 16);
}
__device__ __forceinline__ unsigned pk2bf(float lo, float hi) {
  return (unsigned)f2bf(lo) | ((unsigned)f2bf(hi) << 16);
}

#if defined(__has_builtin)
#if __has_builtin(__builtin_amdgcn_tensor_load_to_lds)
#define HAVE_TDM 1
#endif
#endif

#ifdef HAVE_TDM
// TDM: 16 rows x 32 int32, row stride N_, into LDS with 1-dword pad per 32
// dwords (=> 33-dword LDS pitch, conflict-free column reads).
__device__ __forceinline__ void tdm_load_adj(const int* adj, int elemOff, unsigned ldsOff) {
  unsigned long long ga = (unsigned long long)(const void*)adj + 4ull * (unsigned long long)elemOff;
  v4u g0;
  g0.x = 1u;                                                  // count=1, no gather
  g0.y = ldsOff;                                              // lds_addr (bytes)
  g0.z = (unsigned)ga;                                        // global_addr lo
  g0.w = (unsigned)((ga >> 32) & 0x01FFFFFFull) | (2u << 30); // addr hi | type=2
  v8i g1;
  g1[0] = 0x01120000;   // data_size=4B | pad_enable | pad_interval=32dw | pad=1dw
  g1[1] = 0x04000000;   // tensor_dim0 = 1024
  g1[2] = 0x04000000;   // tensor_dim1 = 1024
  g1[3] = (32 << 16);   // tile_dim0 = 32
  g1[4] = 16;           // tile_dim1 = 16
  g1[5] = 1024;         // tensor_dim0_stride = 1024
  g1[6] = 0; g1[7] = 0;
  v4i z4 = {0, 0, 0, 0};
#if __clang_major__ >= 23
  v8i z8 = {0, 0, 0, 0, 0, 0, 0, 0};
  __builtin_amdgcn_tensor_load_to_lds(g0, g1, z4, z4, z8, 0);
#else
  __builtin_amdgcn_tensor_load_to_lds(g0, g1, z4, z4, 0);
#endif
}
#endif

// ---------------------------------------------------------------------------
// Kernel 0: W -> bf16, pre-swizzled into WMMA B-fragment lane order:
//   wsw[((ct*8 + kb)*32 + lane)*16 + i] = W[(kb*32 + 16*(lane>>4) + i)*256 + ct*16 + (lane&15)]
// ---------------------------------------------------------------------------
__global__ __launch_bounds__(256) void k_wcvt(const float* __restrict__ W,
                                              unsigned short* __restrict__ wsw) {
  int o  = blockIdx.x * 256 + threadIdx.x;
  int i  = o & 15;
  int L  = (o >> 4) & 31;
  int kb = (o >> 9) & 7;
  int ct = o >> 12;
  int src = (kb * 32 + 16 * (L >> 4) + i) * (H_ * F_) + ct * 16 + (L & 15);
  wsw[o] = f2bf(W[src]);
}

// ---------------------------------------------------------------------------
// Kernel 1: g = h @ W (bf16 WMMA, fp32 accum). Writes g fp32 (for s/t) and
// g bf16 pre-swizzled into B-fragment order for the attention GEMM; each
// lane's 8 swizzled outputs are contiguous -> single b128 store.
// ---------------------------------------------------------------------------
__global__ __launch_bounds__(128) void k_gemm_g(const float* __restrict__ h,
                                                const unsigned short* __restrict__ wsw,
                                                float* __restrict__ gf,
                                                unsigned short* __restrict__ gsw) {
  __shared__ unsigned short ah[16][264];
  const int bt   = blockIdx.x;
  const int b    = bt >> 6;
  const int rt   = bt & 63;
  const int tid  = threadIdx.x;
  const int wv   = tid >> 5;
  const int lane = tid & 31;
  const int row0 = rt * 16;

  for (int idx = tid; idx < 16 * 256; idx += 128) {
    int r = idx >> 8, c = idx & 255;
    ah[r][c] = f2bf(h[(b * N_ + row0 + r) * D_ + c]);
  }
  __syncthreads();

  const int ct   = blockIdx.y * 4 + wv;
  const int col0 = ct * 16;
  const int m    = lane & 15;
  const int g2   = lane >> 4;

  v8f acc = {};
#pragma unroll
  for (int kb = 0; kb < 8; ++kb) {
    BFrag a, bfr;
#pragma unroll
    for (int i = 0; i < 16; ++i) {
      int v = i >> 1, hh = i & 1;
      int ka = 16 * g2 + i; (void)v; (void)hh;    // A 16x32 layout, contiguous k = 16*g2 + i
      a.u[i] = ah[m][kb * 32 + ka];
    }
    bfr.v = *(const v16bf*)(wsw + ((ct * 8 + kb) * 32 + lane) * 16);
    acc = __builtin_amdgcn_wmma_f32_16x16x32_bf16(false, a.v, false, bfr.v,
                                                  (short)0, acc, false, false);
  }

  // fp32 g (row-major) for the s/t kernel
#pragma unroll
  for (int r = 0; r < 8; ++r) {
    int row = r + 8 * g2;
    gf[(b * N_ + row0 + row) * D_ + col0 + m] = acc[r];
  }
  // bf16 g in swizzled fragment order: chunk = rt>>1, kg = rt&1 (uniform);
  // lane m owns (kg*16+m); slots i = row&15 = r + 8*g2 are contiguous.
  {
    int swBase = (((b * 16 + ct) * 32 + (rt >> 1)) * 32 + (rt & 1) * 16 + m) * 16;
    uint4 qv;
    qv.x = pk2bf(acc[0], acc[1]);
    qv.y = pk2bf(acc[2], acc[3]);
    qv.z = pk2bf(acc[4], acc[5]);
    qv.w = pk2bf(acc[6], acc[7]);
    *(uint4*)(gsw + swBase + 8 * g2) = qv;       // 16B store, 16B aligned
  }
}

// ---------------------------------------------------------------------------
// Kernel 2: s[b,n,h] = g . Wa[:F],  t[b,n,h] = g . Wa[F:]
// ---------------------------------------------------------------------------
__global__ __launch_bounds__(256) void k_st(const float* __restrict__ gf,
                                            const float* __restrict__ Wa,
                                            float* __restrict__ s,
                                            float* __restrict__ t) {
  int idx = blockIdx.x * 256 + threadIdx.x;
  int hd  = idx & (H_ - 1);
  int bn  = idx >> 3;
  const float* gp = gf + bn * D_ + hd * F_;
  float ss = 0.f, tt = 0.f;
#pragma unroll
  for (int f = 0; f < F_; ++f) {
    float v = gp[f];
    ss += v * Wa[f];
    tt += v * Wa[F_ + f];
  }
  s[idx] = ss;
  t[idx] = tt;
}

// ---------------------------------------------------------------------------
// Kernel 3: fused masked-softmax attention + aggregation (flash style).
// 8 waves = 8 heads; adj tiles double-buffered via TDM one chunk ahead;
// g fragments are single vector loads from the swizzled buffer + prefetch.
// ---------------------------------------------------------------------------
#define APITCH 33
#define ABUF   544
#define CGSTRIDE (32 * 32 * 16)   // ushorts per feature-column group in gsw

__global__ __launch_bounds__(256) void k_attn(const int* __restrict__ adj,
                                              const unsigned short* __restrict__ gsw,
                                              const float* __restrict__ s,
                                              const float* __restrict__ t,
                                              float* __restrict__ out) {
  __shared__ int   adjs[2 * ABUF];
  __shared__ float ts[2][8][32];   // [buf][head][k] -> broadcast reads per wave

  const int bt   = blockIdx.x;
  const int b    = bt >> 6;
  const int it   = bt & 63;
  const int i0   = it * 16;
  const int tid  = threadIdx.x;
  const int hd   = tid >> 5;
  const int lane = tid & 31;
  const int m    = lane & 15;
  const int g2   = lane >> 4;

  const float sI = s[(b * N_ + i0 + m) * H_ + hd];

  float mrun = -1e30f, lrun = 0.f;
  v8f c0 = {}, c1 = {};

  const int rowOff = (b * N_ + i0) * N_;

#ifdef HAVE_TDM
  if (hd == 0)
    tdm_load_adj(adj, rowOff, (unsigned)(unsigned long long)(void*)&adjs[0]);
#else
  for (int idx = tid; idx < 512; idx += 256) {
    int r = idx >> 5, cc = idx & 31;
    adjs[r * APITCH + cc] = adj[rowOff + r * N_ + cc];
  }
#endif
  {
    int jj = tid >> 3, hh2 = tid & 7;
    ts[0][hh2][jj] = t[(b * N_ + jj) * H_ + hh2];
  }

  for (int c = 0; c < N_ / 32; ++c) {
    const int j0  = c * 32;
    const int buf = c & 1;

#ifdef HAVE_TDM
    if (hd == 0) {
      if (c + 1 < N_ / 32) {
        tdm_load_adj(adj, rowOff + j0 + 32,
                     (unsigned)(unsigned long long)(void*)&adjs[(buf ^ 1) * ABUF]);
        __builtin_amdgcn_s_wait_tensorcnt(1);
      } else {
        __builtin_amdgcn_s_wait_tensorcnt(0);
      }
    }
#else
    if (c + 1 < N_ / 32) {
      for (int idx = tid; idx < 512; idx += 256) {
        int r = idx >> 5, cc = idx & 31;
        adjs[(buf ^ 1) * ABUF + r * APITCH + cc] = adj[rowOff + r * N_ + j0 + 32 + cc];
      }
    }
#endif
    if (c + 1 < N_ / 32) {
      int jj = tid >> 3, hh2 = tid & 7;
      ts[buf ^ 1][hh2][jj] = t[(b * N_ + j0 + 32 + jj) * H_ + hh2];
      // prefetch next chunk's g fragments (both feature-column groups)
      const unsigned short* pf =
          gsw + (((b * 16 + hd * 2) * 32 + (c + 1)) * 32 + lane) * 16;
      __builtin_prefetch(pf, 0, 0);
      __builtin_prefetch(pf + CGSTRIDE, 0, 0);
    }
    __syncthreads();

    const int* ab = &adjs[buf * ABUF];

    float e16[16];
    int   msk  = 0;
    float cmax = -1e30f;
#pragma unroll
    for (int i = 0; i < 16; ++i) {
      int v = i >> 1, hb = i & 1;
      int k = 16 * g2 + i; (void)v; (void)hb;
      float e = sI + ts[buf][hd][k];
      e = e >= 0.f ? e : 0.2f * e;                 // LeakyReLU(0.2)
      if (ab[m * APITCH + k] != 0) { msk |= (1 << i); cmax = fmaxf(cmax, e); }
      e16[i] = e;
    }
    cmax = fmaxf(cmax, __shfl_xor(cmax, 16, 32));
    float mnew = fmaxf(mrun, cmax);
    float sc   = __expf(mrun - mnew);

    float ps = 0.f;
    float p16[16];
#pragma unroll
    for (int i = 0; i < 16; ++i) {
      float p = ((msk >> i) & 1) ? __expf(e16[i] - mnew) : 0.f;
      ps += p;
      p16[i] = p;
    }
    BFrag a;
#pragma unroll
    for (int j = 0; j < 8; ++j) a.d[j] = pk2bf(p16[2 * j], p16[2 * j + 1]);

    ps   = ps + __shfl_xor(ps, 16, 32);
    lrun = lrun * sc + ps;
    mrun = mnew;

#pragma unroll
    for (int r = 0; r < 8; ++r) {
      float scr = __shfl(sc, r + 8 * g2, 32);
      c0[r] *= scr;
      c1[r] *= scr;
    }

    // B fragments: single 32B vector loads from swizzled g
    const unsigned short* bp =
        gsw + (((b * 16 + hd * 2) * 32 + c) * 32 + lane) * 16;
    BFrag b0, b1;
    b0.v = *(const v16bf*)bp;
    b1.v = *(const v16bf*)(bp + CGSTRIDE);

    c0 = __builtin_amdgcn_wmma_f32_16x16x32_bf16(false, a.v, false, b0.v,
                                                 (short)0, c0, false, false);
    c1 = __builtin_amdgcn_wmma_f32_16x16x32_bf16(false, a.v, false, b1.v,
                                                 (short)0, c1, false, false);
    __syncthreads();
  }

#pragma unroll
  for (int r = 0; r < 8; ++r) {
    float rl  = __shfl(lrun, r + 8 * g2, 32);
    float inv = (rl != 0.f) ? 1.0f / rl : 0.f;
    int row = r + 8 * g2;
    int oi  = (b * N_ + i0 + row) * D_ + hd * F_ + m;
    out[oi]      = c0[r] * inv;
    out[oi + 16] = c1[r] * inv;
  }
}

// ---------------------------------------------------------------------------
extern "C" void kernel_launch(void* const* d_in, const int* in_sizes, int n_in,
                              void* d_out, int out_size, void* d_ws, size_t ws_size,
                              hipStream_t stream) {
  const float* h   = (const float*)d_in[0];
  const int*   adj = (const int*)d_in[1];
  const float* W   = (const float*)d_in[2];
  const float* Wa  = (const float*)d_in[3];
  float*       out = (float*)d_out;

  // ws: g fp32 (8MB) | g bf16 swizzled (4MB) | s (256KB) | t (256KB) | W bf16 swizzled (128KB)
  float*          g_f32 = (float*)d_ws;
  unsigned short* g_sw  = (unsigned short*)(g_f32 + B_ * N_ * D_);
  float*          s_ws  = (float*)(g_sw + B_ * N_ * D_);
  float*          t_ws  = s_ws + B_ * N_ * H_;
  unsigned short* w_sw  = (unsigned short*)(t_ws + B_ * N_ * H_);

  k_wcvt<<<(D_ * H_ * F_) / 256, 256, 0, stream>>>(W, w_sw);

  dim3 g1(B_ * 64, 4);
  k_gemm_g<<<g1, 128, 0, stream>>>(h, w_sw, g_f32, g_sw);

  k_st<<<(B_ * N_ * H_) / 256, 256, 0, stream>>>(g_f32, Wa, s_ws, t_ws);

  k_attn<<<dim3(B_ * 64), 256, 0, stream>>>(adj, g_sw, s_ws, t_ws, out);
}
